// LongformerCrossEncoder_40114994545338
// MI455X (gfx1250) — matrix operand order repaired and checked by reading, not compiled, not run on hardware
//
#include <hip/hip_runtime.h>
#include <math.h>

// ---------------------------------------------------------------------------
// Model constants
// ---------------------------------------------------------------------------
#define L_   12
#define NH_  12
#define DH_  64
#define H_   768
#define FF_  3072
#define S_   1024
#define B_   4
#define TOK_ (B_*S_)      // 4096 tokens

typedef unsigned short u16;
typedef unsigned int   u32;

typedef __attribute__((ext_vector_type(16))) __bf16 v16bf;
typedef __attribute__((ext_vector_type(8)))  float  v8f;
typedef __attribute__((ext_vector_type(8)))  u32    v8u;

// ---------------------------------------------------------------------------
// CDNA5 async global->LDS copy (ASYNCcnt-tracked), with portable fallback
// ---------------------------------------------------------------------------
#if defined(__AMDGCN__) && __has_builtin(__builtin_amdgcn_global_load_async_to_lds_b128)
#define ATH_ASYNC_LDS 1
#endif

typedef int vi4 __attribute__((vector_size(16)));
typedef __attribute__((address_space(1))) vi4* as1_vi4;
typedef __attribute__((address_space(3))) vi4* as3_vi4;

__device__ __forceinline__ void cp_async_b128(const u16* g, u16* l) {
#ifdef ATH_ASYNC_LDS
  // param0: global (addrspace(1)) int4* -- generic & global addresses are
  // numerically identical, so go through an integer cast (also drops const).
  // param1: LDS (addrspace(3)) pointer -- needs a real addrspacecast.
  __builtin_amdgcn_global_load_async_to_lds_b128(
      (as1_vi4)(size_t)g, (as3_vi4)l, 0, 0);
#else
  *(uint4*)l = *(const uint4*)g;
#endif
}

template <int N>
__device__ __forceinline__ void async_wait() {
#ifdef ATH_ASYNC_LDS
#if __has_builtin(__builtin_amdgcn_s_wait_asynccnt)
  __builtin_amdgcn_s_wait_asynccnt(N);
#else
  asm volatile("s_wait_asynccnt %0" :: "i"(N) : "memory");
#endif
#endif
}

// ---------------------------------------------------------------------------
// Helpers
// ---------------------------------------------------------------------------
__device__ __forceinline__ u16 f2bf(float f) {
  u32 u = __float_as_uint(f);
  u32 r = u + 0x7FFFu + ((u >> 16) & 1u);   // round-to-nearest-even
  return (u16)(r >> 16);
}

// 16-bit A-matrix 16x32 fragment: element-pair p (v16bf elems 2p,2p+1) -> K index
// lanes 0-15: K 0..7 in v0..3, K 16..23 in v4..7 ; lanes 16-31: +8 (ISA 7.12.2)
__device__ __forceinline__ int kOf(int p, int khalf) {
  return (p < 4) ? (khalf * 8 + 2 * p) : (16 + khalf * 8 + 2 * (p - 4));
}

__device__ __forceinline__ float gelu_exact(float x) {
  return 0.5f * x * (1.0f + erff(x * 0.70710678118654752f));
}

__device__ __forceinline__ v8f wmma_bf16(v16bf a, v16bf b, v8f c) {
  // D = A(16x32) * B(32x16) + C, f32 accumulate
  return __builtin_amdgcn_wmma_f32_16x16x32_bf16(false, a, false, b,
                                                 (short)0, c, false, false);
}

__device__ __forceinline__ float blockSum256(float v, volatile float* red) {
  int tid = threadIdx.x;
#pragma unroll
  for (int off = 16; off >= 1; off >>= 1) v += __shfl_xor(v, off, 32);
  __syncthreads();
  if ((tid & 31) == 0) red[tid >> 5] = v;
  __syncthreads();
  if (tid == 0) {
    float s = 0.f;
#pragma unroll
    for (int i = 0; i < 8; ++i) s += red[i];
    red[0] = s;
  }
  __syncthreads();
  return red[0];
}

__device__ __forceinline__ float blockMax256(float v, volatile float* red) {
  int tid = threadIdx.x;
#pragma unroll
  for (int off = 16; off >= 1; off >>= 1) v = fmaxf(v, __shfl_xor(v, off, 32));
  __syncthreads();
  if ((tid & 31) == 0) red[tid >> 5] = v;
  __syncthreads();
  if (tid == 0) {
    float s = red[0];
#pragma unroll
    for (int i = 1; i < 8; ++i) s = fmaxf(s, red[i]);
    red[0] = s;
  }
  __syncthreads();
  return red[0];
}

// ---------------------------------------------------------------------------
// Embedding + LayerNorm : h = LN(WE[id] + PE[s]) ; outputs f32 + bf16
// grid = TOK_, block = 256
// ---------------------------------------------------------------------------
__global__ __launch_bounds__(256) void embed_ln(
    const float* __restrict__ WE, const float* __restrict__ PE,
    const float* __restrict__ g,  const float* __restrict__ be,
    const int*   __restrict__ ids,
    float* __restrict__ h, u16* __restrict__ hb)
{
  __shared__ float red[8];
  const int t   = blockIdx.x;
  const int s   = t & (S_ - 1);
  const int tid = threadIdx.x;
  const size_t wrow = (size_t)ids[t] * H_;
  float x[3]; float sum = 0.f, sq = 0.f;
#pragma unroll
  for (int i = 0; i < 3; ++i) {
    int j = tid + i * 256;
    float v = WE[wrow + j] + PE[(size_t)s * H_ + j];
    x[i] = v; sum += v; sq += v * v;
  }
  float m  = blockSum256(sum, red) * (1.0f / H_);
  float s2 = blockSum256(sq,  red) * (1.0f / H_) - m * m;
  float rs = rsqrtf(s2 + 1e-5f);
#pragma unroll
  for (int i = 0; i < 3; ++i) {
    int j = tid + i * 256;
    float y = (x[i] - m) * rs * g[j] + be[j];
    h [(size_t)t * H_ + j] = y;
    hb[(size_t)t * H_ + j] = f2bf(y);
  }
}

// ---------------------------------------------------------------------------
// Residual + LayerNorm : h = LN(h + f) ; outputs f32 + bf16 (in-place safe)
// ---------------------------------------------------------------------------
__global__ __launch_bounds__(256) void resid_ln(
    const float* __restrict__ hin, const float* __restrict__ f,
    const float* __restrict__ g,   const float* __restrict__ be,
    float* __restrict__ hout, u16* __restrict__ hb)
{
  __shared__ float red[8];
  const int t = blockIdx.x, tid = threadIdx.x;
  float x[3]; float sum = 0.f, sq = 0.f;
#pragma unroll
  for (int i = 0; i < 3; ++i) {
    int j = tid + i * 256;
    float v = hin[(size_t)t * H_ + j] + f[(size_t)t * H_ + j];
    x[i] = v; sum += v; sq += v * v;
  }
  float m  = blockSum256(sum, red) * (1.0f / H_);
  float s2 = blockSum256(sq,  red) * (1.0f / H_) - m * m;
  float rs = rsqrtf(s2 + 1e-5f);
#pragma unroll
  for (int i = 0; i < 3; ++i) {
    int j = tid + i * 256;
    float y = (x[i] - m) * rs * g[j] + be[j];
    hout[(size_t)t * H_ + j] = y;
    hb  [(size_t)t * H_ + j] = f2bf(y);
  }
}

// ---------------------------------------------------------------------------
// WMMA GEMM: C[M,N] = A_bf16[M,K] * W_f32[K,N] + bias ; epilogue per MODE.
// Block tile 128x128x32, 8 waves (4x2), each wave 32x64 (2x4 WMMA accums).
// Software pipeline: A tiles double-buffered in LDS via async global->LDS
// (ASYNCcnt), W tiles register-staged one iteration ahead (LOADcnt) and
// converted fp32->bf16 in-register (weights are read exactly once from HBM,
// which is the bandwidth floor of this model).
// ---------------------------------------------------------------------------
#define BM 128
#define BN 128
#define BK 32
#define LDA 40    // shorts; 80B rows -> 16B-aligned b128 stores, spread banks
#define LDB 136   // shorts; 272B rows -> 16B-aligned, mild 2-way conflicts

enum { OUT_F32 = 0, OUT_BF16 = 1, OUT_BF16_GELU = 2 };

template <int MODE>
__global__ __launch_bounds__(256) void gemm_bf16_wmma(
    const u16*  __restrict__ A, const float* __restrict__ W,
    const float* __restrict__ bias,
    float* __restrict__ Cf, u16* __restrict__ Cb,
    int M, int N, int K)
{
  __shared__ u16 As[2 * BM * LDA];
  __shared__ u16 Ws[BK * LDB];

  const int tid  = threadIdx.x;
  const int lane = tid & 31, wave = tid >> 5;
  const int wm = wave >> 1, wn = wave & 1;          // 4x2 wave grid
  const int rowBlk = blockIdx.y * BM;
  const int colBlk = blockIdx.x * BN;
  const int r = lane & 15, khalf = lane >> 4;

  const v8f vzero = {0.f, 0.f, 0.f, 0.f, 0.f, 0.f, 0.f, 0.f};
  v8f acc[2][4];
#pragma unroll
  for (int mi = 0; mi < 2; ++mi)
#pragma unroll
    for (int ni = 0; ni < 4; ++ni) acc[mi][ni] = vzero;

  // loader mapping
  const int arow = tid >> 1, ahalf = tid & 1;        // A: 16 bf16 per thread
  const int wrow = tid >> 3, wcol = (tid & 7) * 16;  // W: 16 f32 per thread
  const u16*   agBase = A + (size_t)(rowBlk + arow) * K + ahalf * 16;
  const float* wgBase = W + (size_t)wrow * N + colBlk + wcol;
  u16* aDst = &As[arow * LDA + ahalf * 16];

  const int KT = K / BK;
  float4 wreg[4];

  // ---- prologue: stage tile 0 ----
  cp_async_b128(agBase,     aDst);
  cp_async_b128(agBase + 8, aDst + 8);
#pragma unroll
  for (int i = 0; i < 4; ++i) wreg[i] = *(const float4*)(wgBase + 4 * i);

  for (int kt = 0; kt < KT; ++kt) {
    const int cur = kt & 1, nxt = cur ^ 1;

    // ---- convert + store W tile kt (regs were loaded last iteration) ----
    uint4 pk[2];
    u32*  pkw = (u32*)pk;
#pragma unroll
    for (int i = 0; i < 4; ++i) {
      float4 f = wreg[i];
      pkw[2 * i + 0] = (u32)f2bf(f.x) | ((u32)f2bf(f.y) << 16);
      pkw[2 * i + 1] = (u32)f2bf(f.z) | ((u32)f2bf(f.w) << 16);
    }
    *(uint4*)&Ws[wrow * LDB + wcol]     = pk[0];
    *(uint4*)&Ws[wrow * LDB + wcol + 8] = pk[1];

    // ---- issue next tile (A async -> other LDS buffer, W -> regs) ----
    if (kt + 1 < KT) {
      const u16* ag = agBase + (size_t)(kt + 1) * BK;
      u16* dst = aDst + nxt * (BM * LDA);
      cp_async_b128(ag,     dst);
      cp_async_b128(ag + 8, dst + 8);
      const float* wg = wgBase + (size_t)(kt + 1) * BK * N;
#pragma unroll
      for (int i = 0; i < 4; ++i) wreg[i] = *(const float4*)(wg + 4 * i);
      async_wait<2>();   // tile kt complete; tile kt+1 (2 b128s) in flight
    } else {
      async_wait<0>();
    }
    __syncthreads();

    // ---- fragments + 8 WMMAs from buffer `cur` ----
    const u16* asCur = &As[cur * (BM * LDA)];
    v16bf af[2];
#pragma unroll
    for (int mi = 0; mi < 2; ++mi) {
      v8u au;
#pragma unroll
      for (int p = 0; p < 8; ++p)
        au[p] = *(const u32*)&asCur[(wm * 32 + mi * 16 + r) * LDA + kOf(p, khalf)];
      af[mi] = __builtin_bit_cast(v16bf, au);
    }
#pragma unroll
    for (int ni = 0; ni < 4; ++ni) {
      v8u bu;
#pragma unroll
      for (int p = 0; p < 8; ++p)
        bu[p] = (u32)Ws[kOf(p, khalf) * LDB + wn * 64 + ni * 16 + r] | ((u32)Ws[(kOf(p, khalf) + 1) * LDB + wn * 64 + ni * 16 + r] << 16);
      v16bf bfr = __builtin_bit_cast(v16bf, bu);
#pragma unroll
      for (int mi = 0; mi < 2; ++mi)
        acc[mi][ni] = wmma_bf16(af[mi], bfr, acc[mi][ni]);
    }
    __syncthreads();
  }

  // ---- epilogue: bias (+gelu), store f32/bf16 ----
#pragma unroll
  for (int ni = 0; ni < 4; ++ni) {
    int col = colBlk + wn * 64 + ni * 16 + r;
    float bb = bias ? bias[col] : 0.f;
#pragma unroll
    for (int mi = 0; mi < 2; ++mi) {
#pragma unroll
      for (int j = 0; j < 8; ++j) {
        int row = rowBlk + wm * 32 + mi * 16 + j + 8 * khalf;
        float v = acc[mi][ni][j] + bb;
        if (MODE == OUT_BF16_GELU) v = gelu_exact(v);
        if (MODE == OUT_F32) Cf[(size_t)row * N + col] = v;
        else                 Cb[(size_t)row * N + col] = f2bf(v);
      }
    }
  }
}

// ---------------------------------------------------------------------------
// Flash-style sliding-window attention (WMMA).
// grid = (S/128, NH, B), block = 256 (8 waves). Wave handles 16 queries,
// streams the 768-key window in 32-key chunks with online softmax.
// ---------------------------------------------------------------------------
#define KT_S 34   // KT[64][34]  : K^T tile, row = dh
#define VS_S 72   // Vs[32][72]  : V tile,  row = key (16B-aligned rows)
#define PS_S 34   // Ps[wave][16][34] : per-wave P staging

__global__ __launch_bounds__(256) void attn_wmma(
    const u16* __restrict__ qb, const u16* __restrict__ kb,
    const u16* __restrict__ vb, const int* __restrict__ mask,
    u16* __restrict__ ob)
{
  __shared__ u16 KT[64 * KT_S];
  __shared__ u16 Vs[32 * VS_S];
  __shared__ u16 Ps[8][16 * PS_S];

  const int tid = threadIdx.x, lane = tid & 31, wave = tid >> 5;
  const int b = blockIdx.z, nh = blockIdx.y;
  const int qw0 = blockIdx.x * 128;          // first query of workgroup
  const int ablock = qw0 >> 8;               // 256-query attention block
  const int qi_wg = qw0 & 255;               // block-local base
  const int qi0 = qi_wg + wave * 16;         // wave's first block-local query
  const int r = lane & 15, khalf = lane >> 4;

  // ---- preload Q fragments (dh = 0..63 -> 2 k-steps) ----
  v16bf qa[2];
  {
    const size_t qrow = ((size_t)(b * S_ + qw0 + wave * 16 + r)) * H_ + nh * 64;
#pragma unroll
    for (int ks = 0; ks < 2; ++ks) {
      v8u u;
#pragma unroll
      for (int p = 0; p < 8; ++p)
        u[p] = *(const u32*)&qb[qrow + ks * 32 + kOf(p, khalf)];
      qa[ks] = __builtin_bit_cast(v16bf, u);
    }
  }

  const v8f vzero = {0.f, 0.f, 0.f, 0.f, 0.f, 0.f, 0.f, 0.f};
  v8f o[4];
#pragma unroll
  for (int ni = 0; ni < 4; ++ni) o[ni] = vzero;
  float mrow[8], lrow[8];
#pragma unroll
  for (int j = 0; j < 8; ++j) { mrow[j] = -1e30f; lrow[j] = 0.f; }

  const int clo = qi_wg >> 5;
  int chi = (qi_wg + 127 + 512) >> 5; if (chi > 23) chi = 23;

  for (int ck = clo; ck <= chi; ++ck) {
    __syncthreads();
    // ---- cooperative load of 32-key K/V chunk ----
    {
      int keyL = tid & 31;
      int dh0  = (tid >> 5) * 8;
      int kj   = ck * 32 + keyL;
      int gk   = ablock * 256 + kj - 256;          // global key index
      uint4 kd = {0u, 0u, 0u, 0u}, vd = {0u, 0u, 0u, 0u};
      if (gk >= 0 && gk < S_) {
        size_t gi = ((size_t)(b * S_ + gk)) * H_ + nh * 64 + dh0;
        kd = *(const uint4*)&kb[gi];
        vd = *(const uint4*)&vb[gi];
      }
      const u32* kw = (const u32*)&kd;
#pragma unroll
      for (int i = 0; i < 4; ++i) {
        KT[(dh0 + 2 * i + 0) * KT_S + keyL] = (u16)(kw[i] & 0xffffu);
        KT[(dh0 + 2 * i + 1) * KT_S + keyL] = (u16)(kw[i] >> 16);
      }
      *(uint4*)&Vs[keyL * VS_S + dh0] = vd;
    }
    __syncthreads();

    // wave-uniform skip: chunk outside this wave's band union
    bool skip = (ck * 32 + 31 < qi0) || (ck * 32 > qi0 + 527);
    if (!skip) {
      // ---- scores: S(16x32) = Q(16x64) * K^T ----
      v8f s0 = vzero, s1 = vzero;
#pragma unroll
      for (int ks = 0; ks < 2; ++ks) {
        v8u b0, b1;
#pragma unroll
        for (int p = 0; p < 8; ++p) {
          b0[p] = (u32)KT[(ks * 32 + kOf(p, khalf)) * KT_S + r] | ((u32)KT[(ks * 32 + kOf(p, khalf) + 1) * KT_S + r] << 16);
          b1[p] = (u32)KT[(ks * 32 + kOf(p, khalf)) * KT_S + 16 + r] | ((u32)KT[(ks * 32 + kOf(p, khalf) + 1) * KT_S + 16 + r] << 16);
        }
        s0 = wmma_bf16(qa[ks], __builtin_bit_cast(v16bf, b0), s0);
        s1 = wmma_bf16(qa[ks], __builtin_bit_cast(v16bf, b1), s1);
      }
      // ---- masked online softmax (16-lane row reductions) ----
#pragma unroll
      for (int j = 0; j < 8; ++j) {
        int qi = qi0 + j + 8 * khalf;
        float sv[2];
#pragma unroll
        for (int g = 0; g < 2; ++g) {
          int kj = ck * 32 + g * 16 + r;
          int gk = ablock * 256 + kj - 256;
          int d  = kj - qi;
          bool ok = (d >= 0) && (d <= 512) && (gk >= 0) && (gk < S_) &&
                    (mask[b * S_ + gk] > 0);
          float raw = (g ? s1[j] : s0[j]) * 0.125f;  // 1/sqrt(64)
          sv[g] = ok ? raw : -1e9f;
        }
        float vmax = fmaxf(sv[0], sv[1]);
#pragma unroll
        for (int off = 8; off >= 1; off >>= 1)
          vmax = fmaxf(vmax, __shfl_xor(vmax, off, 16));
        float mnew = fmaxf(mrow[j], vmax);
        float fac  = expf(mrow[j] - mnew);
        float p0   = expf(sv[0] - mnew);
        float p1   = expf(sv[1] - mnew);
        float ps   = p0 + p1;
#pragma unroll
        for (int off = 8; off >= 1; off >>= 1) ps += __shfl_xor(ps, off, 16);
        lrow[j] = lrow[j] * fac + ps;
        mrow[j] = mnew;
#pragma unroll
        for (int ni = 0; ni < 4; ++ni) o[ni][j] *= fac;
        int rr = j + 8 * khalf;
        Ps[wave][rr * PS_S + r]      = f2bf(p0);
        Ps[wave][rr * PS_S + 16 + r] = f2bf(p1);
      }
      // ---- O += P(16x32) * V(32x64) : 4 WMMAs ----
      v8u pu;
#pragma unroll
      for (int p = 0; p < 8; ++p)
        pu[p] = *(const u32*)&Ps[wave][r * PS_S + kOf(p, khalf)];
      v16bf pa = __builtin_bit_cast(v16bf, pu);
#pragma unroll
      for (int ni = 0; ni < 4; ++ni) {
        v8u vu;
#pragma unroll
        for (int p = 0; p < 8; ++p)
          vu[p] = (u32)Vs[kOf(p, khalf) * VS_S + ni * 16 + r] | ((u32)Vs[(kOf(p, khalf) + 1) * VS_S + ni * 16 + r] << 16);
        o[ni] = wmma_bf16(pa, __builtin_bit_cast(v16bf, vu), o[ni]);
      }
    }
  }

  // ---- normalize + store (bf16, feeds the O-projection GEMM) ----
#pragma unroll
  for (int ni = 0; ni < 4; ++ni) {
#pragma unroll
    for (int j = 0; j < 8; ++j) {
      int row = qw0 + wave * 16 + j + 8 * khalf;
      float ov = o[ni][j] / lrow[j];
      ob[((size_t)(b * S_ + row)) * H_ + nh * 64 + ni * 16 + r] = f2bf(ov);
    }
  }
}

// ---------------------------------------------------------------------------
// Attention pooling head: out[b] = sum_s softmax(h.aw)_s * (h_s . Wc) + bc
// grid = B, block = 256
// ---------------------------------------------------------------------------
__global__ __launch_bounds__(256) void pool_head(
    const float* __restrict__ h, const float* __restrict__ aw,
    const float* __restrict__ Wc, const float* __restrict__ bc,
    float* __restrict__ outp)
{
  __shared__ float sc[S_];
  __shared__ float red[8];
  const int b = blockIdx.x, tid = threadIdx.x;

  for (int s = tid; s < S_; s += 256) {
    const float* hp = h + ((size_t)(b * S_ + s)) * H_;
    float d = 0.f;
    for (int j = 0; j < H_; ++j) d += hp[j] * aw[j];
    sc[s] = d;
  }
  __syncthreads();
  float mx = -1e30f;
  for (int s = tid; s < S_; s += 256) mx = fmaxf(mx, sc[s]);
  mx = blockMax256(mx, red);
  float sum = 0.f;
  for (int s = tid; s < S_; s += 256) { float e = expf(sc[s] - mx); sc[s] = e; sum += e; }
  sum = blockSum256(sum, red);
  float inv = 1.f / sum;
  __syncthreads();
  float acc = 0.f;
  for (int s = tid; s < S_; s += 256) {
    const float* hp = h + ((size_t)(b * S_ + s)) * H_;
    float d = 0.f;
    for (int j = 0; j < H_; ++j) d += hp[j] * Wc[j];
    acc += sc[s] * inv * d;
  }
  acc = blockSum256(acc, red);
  if (tid == 0) outp[b] = acc + bc[0];
}

// ---------------------------------------------------------------------------
// Orchestration
// ---------------------------------------------------------------------------
extern "C" void kernel_launch(void* const* d_in, const int* in_sizes, int n_in,
                              void* d_out, int out_size, void* d_ws, size_t ws_size,
                              hipStream_t stream)
{
  (void)in_sizes; (void)n_in; (void)out_size; (void)ws_size;
  const float* WE    = (const float*)d_in[0];
  const float* PE    = (const float*)d_in[1];
  const float* Eg    = (const float*)d_in[2];
  const float* Eb    = (const float*)d_in[3];
  const float* Wq    = (const float*)d_in[4];
  const float* bq    = (const float*)d_in[5];
  const float* Wk    = (const float*)d_in[6];
  const float* bk    = (const float*)d_in[7];
  const float* Wv    = (const float*)d_in[8];
  const float* bv    = (const float*)d_in[9];
  const float* Wo    = (const float*)d_in[10];
  const float* bo    = (const float*)d_in[11];
  const float* g1    = (const float*)d_in[12];
  const float* beta1 = (const float*)d_in[13];
  const float* W1    = (const float*)d_in[14];
  const float* b1    = (const float*)d_in[15];
  const float* W2    = (const float*)d_in[16];
  const float* b2    = (const float*)d_in[17];
  const float* g2    = (const float*)d_in[18];
  const float* beta2 = (const float*)d_in[19];
  const float* aw    = (const float*)d_in[20];
  const float* Wc    = (const float*)d_in[21];
  const float* bc    = (const float*)d_in[22];
  const int*   ids   = (const int*)d_in[23];
  const int*   amask = (const int*)d_in[24];

  char* p = (char*)d_ws;
  auto alloc = [&](size_t bytes) {
    char* q = p; p += (bytes + 255) & ~(size_t)255; return q;
  };
  float* h    = (float*)alloc((size_t)TOK_ * H_ * 4);
  u16*   hb   = (u16*)  alloc((size_t)TOK_ * H_ * 2);
  u16*   qbuf = (u16*)  alloc((size_t)TOK_ * H_ * 2);
  u16*   kbuf = (u16*)  alloc((size_t)TOK_ * H_ * 2);
  u16*   vbuf = (u16*)  alloc((size_t)TOK_ * H_ * 2);
  u16*   abuf = (u16*)  alloc((size_t)TOK_ * H_ * 2);
  float* t1   = (float*)alloc((size_t)TOK_ * H_ * 4);
  u16*   midb = (u16*)  alloc((size_t)TOK_ * FF_ * 2);

  const dim3 blk(256);
  const dim3 gH(H_ / BN, TOK_ / BM);    // 6 x 32
  const dim3 gF(FF_ / BN, TOK_ / BM);   // 24 x 32
  const dim3 gA(S_ / 128, NH_, B_);     // 8 x 12 x 4

  embed_ln<<<TOK_, blk, 0, stream>>>(WE, PE, Eg, Eb, ids, h, hb);

  for (int l = 0; l < L_; ++l) {
    const size_t oHH = (size_t)l * H_ * H_;
    const size_t oHF = (size_t)l * H_ * FF_;
    const size_t oH  = (size_t)l * H_;
    const size_t oF  = (size_t)l * FF_;

    gemm_bf16_wmma<OUT_BF16><<<gH, blk, 0, stream>>>(
        hb, Wq + oHH, bq + oH, nullptr, qbuf, TOK_, H_, H_);
    gemm_bf16_wmma<OUT_BF16><<<gH, blk, 0, stream>>>(
        hb, Wk + oHH, bk + oH, nullptr, kbuf, TOK_, H_, H_);
    gemm_bf16_wmma<OUT_BF16><<<gH, blk, 0, stream>>>(
        hb, Wv + oHH, bv + oH, nullptr, vbuf, TOK_, H_, H_);

    attn_wmma<<<gA, blk, 0, stream>>>(qbuf, kbuf, vbuf, amask, abuf);

    gemm_bf16_wmma<OUT_F32><<<gH, blk, 0, stream>>>(
        abuf, Wo + oHH, bo + oH, t1, nullptr, TOK_, H_, H_);
    resid_ln<<<TOK_, blk, 0, stream>>>(h, t1, g1 + oH, beta1 + oH, h, hb);

    gemm_bf16_wmma<OUT_BF16_GELU><<<gF, blk, 0, stream>>>(
        hb, W1 + oHF, b1 + oF, nullptr, midb, TOK_, FF_, H_);
    gemm_bf16_wmma<OUT_F32><<<gH, blk, 0, stream>>>(
        midb, W2 + oHF, b2 + oH, t1, nullptr, TOK_, H_, FF_);
    resid_ln<<<TOK_, blk, 0, stream>>>(h, t1, g2 + oH, beta2 + oH, h, hb);
  }

  pool_head<<<B_, blk, 0, stream>>>(h, aw, Wc, bc, (float*)d_out);
}
